// SpikeFP8Adder_Spatial_23407571764129
// MI455X (gfx1250) — compile-verified, matrix operand order checked by reading, not run on hardware
//
#include <hip/hip_runtime.h>

// 12-bit ripple-carry adder on {0,1} float "spikes".
// Per wave32: pack 16 rows' 24 input bits into integer sums via one
// V_WMMA_F32_16X16X32_F16 dot with power-of-two weights (exact in f16/f32),
// then a fully branchless VALU unpack of the 13-bit results.
// Bandwidth-bound: ~620 MB compulsory traffic -> ~27 us floor at 23.3 TB/s.

typedef __attribute__((ext_vector_type(16))) _Float16 v16h;
typedef __attribute__((ext_vector_type(8)))  float    v8f;
typedef __attribute__((ext_vector_type(4)))  float    v4f;

__global__ __launch_bounds__(256) void snn_adder12_wmma_kernel(
    const float* __restrict__ A, const float* __restrict__ B,
    float* __restrict__ out, int n_rows, int n_tiles)
{
    const int lane = threadIdx.x & 31;
    const int m    = lane & 15;        // row within half-tile / output column
    const int half = lane >> 4;        // A-fragment K-half selector
    const int q    = m;                // output-bit assignment for unpack

    const int wave_global = blockIdx.x * 8 + (threadIdx.x >> 5); // 256 thr = 8 waves
    const int n_waves     = gridDim.x * 8;

    // Constant B fragment: w[k] = 2^(11-k) for k<12 (A bits, MSB first),
    // w[12+j] = 2^(11-j) (B bits), w[24..31] = 0 (kills the dummy loads below).
    // 16-bit B 32x16 striping mirrors A's K layout across lane halves.
    v16h bfrag;
    {
        const _Float16 w_lo[16] = { 2048, 1024, 512, 256, 128, 64, 32, 16,   // K0..7
                                     128,   64,  32,  16,   8,  4,  2,  1 }; // K16..23
        const _Float16 w_hi[16] = {    8,    4,   2,   1, 2048,1024,512,256, // K8..15
                                       0,    0,   0,   0,   0,  0,  0,  0 }; // K24..31
#pragma unroll
        for (int i = 0; i < 16; ++i)
            bfrag[i] = half ? w_hi[i] : w_lo[i];
    }

    // Lane-constant gather bases (hoisted; no per-tile pointer selects).
    // half==0: K0..7 = A bits 0..7, K16..23 = B bits 4..11
    // half==1: K8..11 = A bits 8..11, K12..15 = B bits 0..3,
    //          K24..31 slots load valid dummies (zeroed by bfrag weights).
    const float* const base0 = half ? A + 8 : A + 0;
    const float* const base1 = half ? B + 0 : A + 4;
    const float* const base2 = half ? A + 0 : B + 4;   // dummy when half==1
    const float* const base3 = half ? B + 4 : B + 8;   // dummy when half==1

    // Per-lane unpack constants (hoisted; no branching inside the tile loop).
    // q<12  : write sum bit q of 8 rows, row stride 12 floats (48B).
    // q>=12 : write carry of 8 rows, row stride 1 float (q=13..15 duplicate
    //         q=12 byte-for-byte -> benign, keeps EXEC full / zero divergence).
    const bool     is_sum   = (q < 12);
    const int      shift    = is_sum ? (11 - q) : 12;
    const unsigned stride_b = is_sum ? 48u : 4u;        // byte stride per row

    for (int tile = wave_global; tile < n_tiles; tile += n_waves) {
        const int row0 = tile * 16;
        const int row  = row0 + m;
        const unsigned roff = (unsigned)row * 12u;       // element offset of row

        // Prefetch next grid-stride tile (global_prefetch, NT path).
        if (tile + n_waves < n_tiles) {
            const unsigned adv = roff + (unsigned)n_waves * 16u * 12u;
            __builtin_prefetch(A + adv, 0, 0);
            __builtin_prefetch(B + adv, 0, 0);
        }

        // Branchless fragment gather: 4 NT b128 loads (48B row stride, 16B aligned).
        const v4f L0 = __builtin_nontemporal_load((const v4f*)(base0 + roff));
        const v4f L1 = __builtin_nontemporal_load((const v4f*)(base1 + roff));
        const v4f L2 = __builtin_nontemporal_load((const v4f*)(base2 + roff));
        const v4f L3 = __builtin_nontemporal_load((const v4f*)(base3 + roff));

        v16h a;
#pragma unroll
        for (int i = 0; i < 4; ++i) {
            a[i + 0]  = (_Float16)L0[i];
            a[i + 4]  = (_Float16)L1[i];
            a[i + 8]  = (_Float16)L2[i];
            a[i + 12] = (_Float16)L3[i];
        }

        // D[row][*] = integer sum a+b of that row, replicated across 16 columns.
        v8f c = {};
        c = __builtin_amdgcn_wmma_f32_16x16x32_f16(
                /*neg_a=*/false, a, /*neg_b=*/false, bfrag,
                /*c_mod=*/(short)0, c, /*reuse_a=*/false, /*reuse_b=*/false);

        // Branchless unpack: lane holds rows (half*8 .. half*8+7) in c[0..7].
        // 32-bit offset select (max offset n_rows*13 < 2^31) -> v_cndmask, no branch.
        const unsigned rbase = (unsigned)row0 + (unsigned)half * 8u;
        const unsigned off_e = is_sum ? (rbase * 12u + (unsigned)q)
                                      : ((unsigned)n_rows * 12u + rbase);
        unsigned boff = off_e * 4u;                      // byte offset from out
#pragma unroll
        for (int j = 0; j < 8; ++j) {
            const unsigned u   = (unsigned)c[j];         // exact integer 0..8190
            const float    bit = (float)((u >> shift) & 1u);
            __builtin_nontemporal_store(bit, (float*)((char*)out + boff));
            boff += stride_b;
        }
    }
}

// Scalar tail for rows not covered by 16-row tiles (unused for N = 2^22).
__global__ void snn_adder12_tail_kernel(
    const float* __restrict__ A, const float* __restrict__ B,
    float* __restrict__ out, int n_rows, int row_start)
{
    int r = row_start + blockIdx.x * blockDim.x + threadIdx.x;
    if (r >= n_rows) return;
    unsigned a = 0, b = 0;
#pragma unroll
    for (int i = 0; i < 12; ++i) {
        a = (a << 1) | (A[(size_t)r * 12 + i] != 0.0f ? 1u : 0u);
        b = (b << 1) | (B[(size_t)r * 12 + i] != 0.0f ? 1u : 0u);
    }
    const unsigned u = a + b;
#pragma unroll
    for (int i = 0; i < 12; ++i)
        out[(size_t)r * 12 + i] = (float)((u >> (11 - i)) & 1u);
    out[(size_t)n_rows * 12 + r] = (float)((u >> 12) & 1u);
}

extern "C" void kernel_launch(void* const* d_in, const int* in_sizes, int n_in,
                              void* d_out, int out_size, void* d_ws, size_t ws_size,
                              hipStream_t stream) {
    const float* A = (const float*)d_in[0];
    const float* B = (const float*)d_in[1];
    float* out = (float*)d_out;

    const int n_rows  = in_sizes[0] / 12;
    const int n_tiles = n_rows / 16;

    if (n_tiles > 0) {
        // 4096 blocks x 8 waves = 32768 waves; 262144 tiles -> 8 tiles/wave
        // via grid-stride loop (amortizes setup, enables next-tile prefetch).
        int blocks = 4096;
        int waves  = blocks * 8;
        if (n_tiles < waves) blocks = (n_tiles + 7) / 8;
        snn_adder12_wmma_kernel<<<blocks, 256, 0, stream>>>(A, B, out, n_rows, n_tiles);
    }

    const int row_start = n_tiles * 16;
    const int rem = n_rows - row_start;
    if (rem > 0) {
        snn_adder12_tail_kernel<<<(rem + 255) / 256, 256, 0, stream>>>(
            A, B, out, n_rows, row_start);
    }
}
